// SConvLayer_10058813407856
// MI455X (gfx1250) — matrix operand order repaired and checked by reading, not compiled, not run on hardware
//
#include <hip/hip_runtime.h>
#include <hip/hip_bf16.h>

typedef __attribute__((ext_vector_type(16))) _Float16 v16h;
typedef __attribute__((ext_vector_type(8)))  _Float16 v8h;
typedef __attribute__((ext_vector_type(4)))  _Float16 v4h;
typedef __attribute__((ext_vector_type(8)))  float    v8f;

#define BB 4
#define LL 4096
#define DD 1024
#define MT (BB * LL)          // 16384 rows

#define BM 128
#define BN 64
#define BK 32
#define APITCH 40             // 32 halfs + 8 pad (80B pitch, 16B aligned)

#define NCH 64                // scan chunks
#define CLEN (LL / NCH)       // 64 steps per chunk

__device__ __forceinline__ float silu_f(float v) {
    return v / (1.0f + __expf(-v));
}

__device__ __forceinline__ void wait_async0() {
#if __has_builtin(__builtin_amdgcn_s_wait_asynccnt)
    __builtin_amdgcn_s_wait_asynccnt(0);
#else
    asm volatile("s_wait_asynccnt 0" ::: "memory");
#endif
}

// async copy of 16 bytes/lane from global (saddr + 32-bit voffset) into LDS
__device__ __forceinline__ void async_b128(unsigned lds_byte_addr,
                                           unsigned gvoff,
                                           const _Float16* gbase) {
    asm volatile("global_load_async_to_lds_b128 %0, %1, %2"
                 :
                 : "v"(lds_byte_addr), "v"(gvoff), "s"(gbase)
                 : "memory");
}

#define SHUF16(lo, hi) __builtin_shufflevector((lo), (hi), 0,1,2,3,4,5,6,7,8,9,10,11,12,13,14,15)

// ---------------------------------------------------------------------------
// Kernel 0: f32 -> f16 conversion (x and the four weight matrices)
// ---------------------------------------------------------------------------
__global__ __launch_bounds__(256) void cvt_kernel(
    const float* __restrict__ s, _Float16* __restrict__ dst, int n)
{
    const int i = (blockIdx.x * 256 + threadIdx.x) * 8;
    if (i >= n) return;
    const float4 a = *reinterpret_cast<const float4*>(s + i);
    const float4 b = *reinterpret_cast<const float4*>(s + i + 4);
    v8h o = { (_Float16)a.x, (_Float16)a.y, (_Float16)a.z, (_Float16)a.w,
              (_Float16)b.x, (_Float16)b.y, (_Float16)b.z, (_Float16)b.w };
    *reinterpret_cast<v8h*>(dst + i) = o;
}

// ---------------------------------------------------------------------------
// Kernel 1: z = (x@Wz^T + bz) * silu(x@Wza^T + bza)
//   f16 WMMA, f32 accum; async double-buffered LDS staging
// ---------------------------------------------------------------------------
__global__ __launch_bounds__(256) void proj_z_kernel(
    const _Float16* __restrict__ xh,
    const _Float16* __restrict__ Wzh,  const float* __restrict__ bz,
    const _Float16* __restrict__ Wzah, const float* __restrict__ bza,
    float* __restrict__ zbuf)
{
    __shared__ _Float16 As[2][BM * APITCH];
    __shared__ _Float16 Bs[2][2][BN * APITCH];

    const int tid    = threadIdx.x;
    const int m0     = blockIdx.y * BM;
    const int n0     = blockIdx.x * BN;
    const int wave   = tid >> 5;
    const int lane   = tid & 31;
    const int laneM  = lane & 15;
    const int laneHi = lane >> 4;
    const int wm     = wave >> 1;   // 0..3
    const int wn     = wave & 1;    // 0..1

    const v8f vzero = {0.f,0.f,0.f,0.f,0.f,0.f,0.f,0.f};
    v8f acc[2][2][2];
    #pragma unroll
    for (int w = 0; w < 2; ++w)
        #pragma unroll
        for (int i = 0; i < 2; ++i)
            #pragma unroll
            for (int j = 0; j < 2; ++j) acc[w][i][j] = vzero;

    // ---- async tile issue: A is 128x32 halfs (512 16B chunks, 2/thread),
    //      each B is 64x32 halfs (256 16B chunks, 1/thread)
    auto issue_tiles = [&](int buf, int k0) {
        #pragma unroll
        for (int r = 0; r < 2; ++r) {
            const int q   = tid + r * 256;
            const int row = q >> 2;
            const int c8  = (q & 3) * 8;
            const unsigned lds = (unsigned)(size_t)&As[buf][row * APITCH + c8];
            const unsigned go  = (unsigned)(((m0 + row) * DD + k0 + c8) * 2);
            async_b128(lds, go, xh);
        }
        {
            const int row = tid >> 2;
            const int c8  = (tid & 3) * 8;
            const unsigned go = (unsigned)(((n0 + row) * DD + k0 + c8) * 2);
            async_b128((unsigned)(size_t)&Bs[buf][0][row * APITCH + c8], go, Wzh);
            async_b128((unsigned)(size_t)&Bs[buf][1][row * APITCH + c8], go, Wzah);
        }
    };

    const int NK = DD / BK;     // 32
    issue_tiles(0, 0);

    for (int kk = 0; kk < NK; ++kk) {
        const int cur = kk & 1;
        wait_async0();
        __syncthreads();                       // buf `cur` ready for everyone
        if (kk + 1 < NK) issue_tiles(1 - cur, (kk + 1) * BK);

        #pragma unroll
        for (int ti = 0; ti < 2; ++ti) {
            const int mrow = wm * 32 + ti * 16 + laneM;
            const int kc0  = laneHi ? 8 : 0;   // A lane K layout per ISA table
            v8h alo = *reinterpret_cast<const v8h*>(&As[cur][mrow * APITCH + kc0]);
            v8h ahi = *reinterpret_cast<const v8h*>(&As[cur][mrow * APITCH + kc0 + 16]);
            v16h a  = SHUF16(alo, ahi);
            #pragma unroll
            for (int w = 0; w < 2; ++w) {
                #pragma unroll
                for (int tj = 0; tj < 2; ++tj) {
                    const int nrow = wn * 32 + tj * 16 + laneM;
                    const int kb   = laneHi * 8;    // B lane K layout per ISA table (matches A)
                    v8h b0 = *reinterpret_cast<const v8h*>(&Bs[cur][w][nrow * APITCH + kb]);
                    v8h b1 = *reinterpret_cast<const v8h*>(&Bs[cur][w][nrow * APITCH + kb + 16]);
                    v16h b = SHUF16(b0, b1);
                    acc[w][ti][tj] = __builtin_amdgcn_wmma_f32_16x16x32_f16(
                        false, a, false, b, (short)0, acc[w][ti][tj], false, false);
                }
            }
        }
        __syncthreads();                       // all reads of `cur` done before overwrite
    }

    #pragma unroll
    for (int ti = 0; ti < 2; ++ti) {
        #pragma unroll
        for (int tj = 0; tj < 2; ++tj) {
            const int nn  = n0 + wn * 32 + tj * 16 + laneM;
            const float b0 = bz[nn];
            const float b1 = bza[nn];
            #pragma unroll
            for (int r = 0; r < 8; ++r) {
                const int mm = m0 + wm * 32 + ti * 16 + laneHi * 8 + r;
                const float vz  = acc[0][ti][tj][r] + b0;
                const float vza = acc[1][ti][tj][r] + b1;
                zbuf[(size_t)mm * DD + nn] = vz * silu_f(vza);
            }
        }
    }
}

// ---------------------------------------------------------------------------
// Scan (3-pass chunked): h[l] = z[l] + p*h[l-1], p = exp(1e-5 + i*ang(d)),
// h[-1] = lc.  Pass A: chunk-local state with zero init.
// ---------------------------------------------------------------------------
__device__ __forceinline__ float chan_angle(int d) {
    return __expf(__logf(0.001f) * ((float)d / (float)(DD - 1)));
}

__global__ __launch_bounds__(256) void scan_local_kernel(
    const float* __restrict__ z, float* __restrict__ carR, float* __restrict__ carI)
{
    const int tid = blockIdx.x * 256 + threadIdx.x;   // 0 .. B*NCH*D-1
    const int d   = tid % DD;
    const int bc  = tid / DD;                         // b*NCH + c
    const int b   = bc / NCH;
    const int c   = bc % NCH;

    const float ang = chan_angle(d);
    const float er_ = __expf(1e-5f);
    const float pr  = er_ * __cosf(ang);
    const float pi  = er_ * __sinf(ang);

    float er = 0.0f, ei = 0.0f;
    const float* p = z + ((size_t)b * LL + (size_t)c * CLEN) * DD + d;
    #pragma unroll 8
    for (int t = 0; t < CLEN; ++t) {
        const float zv = p[(size_t)t * DD];
        const float nr = __fmaf_rn(pr, er, __fmaf_rn(-pi, ei, zv));
        const float ni = __fmaf_rn(pr, ei, pi * er);
        er = nr; ei = ni;
    }
    carR[(size_t)bc * DD + d] = er;
    carI[(size_t)bc * DD + d] = ei;
}

// Pass B: sequential combine across chunks; replaces E_c with the carry-in
// state for chunk c (i.e. h at index c*CLEN-1).
__global__ __launch_bounds__(256) void scan_carry_kernel(
    float* __restrict__ carR, float* __restrict__ carI,
    const float* __restrict__ lc_re, const float* __restrict__ lc_im)
{
    const int tid = blockIdx.x * 256 + threadIdx.x;   // 0 .. B*D-1
    const int b = tid / DD;
    const int d = tid % DD;

    const float ang = chan_angle(d);
    const float prc = __expf((float)CLEN * 1e-5f) * __cosf((float)CLEN * ang);
    const float pic = __expf((float)CLEN * 1e-5f) * __sinf((float)CLEN * ang);

    float cr = lc_re[d];
    float ci = lc_im[d];
    for (int c = 0; c < NCH; ++c) {
        const size_t idx = ((size_t)b * NCH + c) * DD + d;
        const float er = carR[idx];
        const float ei = carI[idx];
        carR[idx] = cr;                    // carry-in for chunk c
        carI[idx] = ci;
        const float nr = er + prc * cr - pic * ci;   // state at end of chunk c
        const float ni = ei + prc * ci + pic * cr;
        cr = nr; ci = ni;
    }
}

// Pass C: apply carries, write Re(h) in place over z.
__global__ __launch_bounds__(256) void scan_apply_kernel(
    float* __restrict__ zh,
    const float* __restrict__ carR, const float* __restrict__ carI)
{
    const int tid = blockIdx.x * 256 + threadIdx.x;
    const int d   = tid % DD;
    const int bc  = tid / DD;
    const int b   = bc / NCH;
    const int c   = bc % NCH;

    const float ang = chan_angle(d);
    const float er_ = __expf(1e-5f);
    const float pr  = er_ * __cosf(ang);
    const float pi  = er_ * __sinf(ang);

    float hr = carR[(size_t)bc * DD + d];
    float hi = carI[(size_t)bc * DD + d];
    float* p = zh + ((size_t)b * LL + (size_t)c * CLEN) * DD + d;
    #pragma unroll 8
    for (int t = 0; t < CLEN; ++t) {
        const float zv = p[(size_t)t * DD];
        const float nr = __fmaf_rn(pr, hr, __fmaf_rn(-pi, hi, zv));
        const float ni = __fmaf_rn(pr, hi, pi * hr);
        p[(size_t)t * DD] = nr;
        hr = nr; hi = ni;
    }
}

// ---------------------------------------------------------------------------
// inv_rms[m] = rsqrt(mean(h[m]^2) + 1e-6), one wave per row
// ---------------------------------------------------------------------------
__global__ __launch_bounds__(256) void rms_kernel(
    const float* __restrict__ h, float* __restrict__ inv_rms)
{
    const int row  = blockIdx.x * 8 + (threadIdx.x >> 5);
    const int lane = threadIdx.x & 31;
    const float* r = h + (size_t)row * DD;
    float s = 0.0f;
    #pragma unroll
    for (int i = 0; i < 8; ++i) {
        const float4 v = *reinterpret_cast<const float4*>(r + lane * 4 + i * 128);
        s += v.x * v.x + v.y * v.y + v.z * v.z + v.w * v.w;
    }
    #pragma unroll
    for (int off = 16; off > 0; off >>= 1) s += __shfl_xor(s, off, 32);
    if (lane == 0) inv_rms[row] = rsqrtf(s * (1.0f / DD) + 1e-6f);
}

// ---------------------------------------------------------------------------
// Kernel 4: y = (normed@Wy^T + by) * silu(x@Wya^T + bya)
//   A1 = h*inv_rms*norm_w (VGPR-staged f32->f16), A2 = xh (async),
//   B = Wyh/Wyah (async); double-buffered
// ---------------------------------------------------------------------------
__global__ __launch_bounds__(256) void out_gemm_kernel(
    const float* __restrict__ h,  const _Float16* __restrict__ xh,
    const _Float16* __restrict__ Wyh, const float* __restrict__ by,
    const _Float16* __restrict__ Wyah, const float* __restrict__ bya,
    const float* __restrict__ norm_w, const float* __restrict__ inv_rms,
    float* __restrict__ y)
{
    __shared__ _Float16 As1[2][BM * APITCH];   // normed
    __shared__ _Float16 As2[2][BM * APITCH];   // x
    __shared__ _Float16 Bs[2][2][BN * APITCH]; // Wy, Wya

    const int tid    = threadIdx.x;
    const int m0     = blockIdx.y * BM;
    const int n0     = blockIdx.x * BN;
    const int wave   = tid >> 5;
    const int lane   = tid & 31;
    const int laneM  = lane & 15;
    const int laneHi = lane >> 4;
    const int wm     = wave >> 1;
    const int wn     = wave & 1;

    const v8f vzero = {0.f,0.f,0.f,0.f,0.f,0.f,0.f,0.f};
    v8f acc[2][2][2];
    #pragma unroll
    for (int w = 0; w < 2; ++w)
        #pragma unroll
        for (int i = 0; i < 2; ++i)
            #pragma unroll
            for (int j = 0; j < 2; ++j) acc[w][i][j] = vzero;

    auto issue_tiles = [&](int buf, int k0) {
        // A2 + B via async copies
        #pragma unroll
        for (int r = 0; r < 2; ++r) {
            const int q   = tid + r * 256;
            const int row = q >> 2;
            const int c8  = (q & 3) * 8;
            const unsigned go = (unsigned)(((m0 + row) * DD + k0 + c8) * 2);
            async_b128((unsigned)(size_t)&As2[buf][row * APITCH + c8], go, xh);
        }
        {
            const int row = tid >> 2;
            const int c8  = (tid & 3) * 8;
            const unsigned go = (unsigned)(((n0 + row) * DD + k0 + c8) * 2);
            async_b128((unsigned)(size_t)&Bs[buf][0][row * APITCH + c8], go, Wyh);
            async_b128((unsigned)(size_t)&Bs[buf][1][row * APITCH + c8], go, Wyah);
        }
        // A1: normed = h * inv_rms[m] * norm_w[k], converted f32->f16 in VGPRs
        #pragma unroll
        for (int r = 0; r < 4; ++r) {
            const int q   = tid + r * 256;
            const int row = q >> 3;
            const int c4  = (q & 7) << 2;
            const size_t gi = (size_t)(m0 + row) * DD + k0 + c4;
            const float irm = inv_rms[m0 + row];
            const float4 vh = *reinterpret_cast<const float4*>(h + gi);
            const float4 nw = *reinterpret_cast<const float4*>(norm_w + k0 + c4);
            v4h h1 = { (_Float16)(vh.x * irm * nw.x), (_Float16)(vh.y * irm * nw.y),
                       (_Float16)(vh.z * irm * nw.z), (_Float16)(vh.w * irm * nw.w) };
            *reinterpret_cast<v4h*>(&As1[buf][row * APITCH + c4]) = h1;
        }
    };

    const int NK = DD / BK;
    issue_tiles(0, 0);

    for (int kk = 0; kk < NK; ++kk) {
        const int cur = kk & 1;
        wait_async0();
        __syncthreads();          // drains DScnt (A1 stores) + async, all waves
        if (kk + 1 < NK) issue_tiles(1 - cur, (kk + 1) * BK);

        #pragma unroll
        for (int ti = 0; ti < 2; ++ti) {
            const int mrow = wm * 32 + ti * 16 + laneM;
            const int kc0  = laneHi ? 8 : 0;
            v8h a1lo = *reinterpret_cast<const v8h*>(&As1[cur][mrow * APITCH + kc0]);
            v8h a1hi = *reinterpret_cast<const v8h*>(&As1[cur][mrow * APITCH + kc0 + 16]);
            v16h a1  = SHUF16(a1lo, a1hi);
            v8h a2lo = *reinterpret_cast<const v8h*>(&As2[cur][mrow * APITCH + kc0]);
            v8h a2hi = *reinterpret_cast<const v8h*>(&As2[cur][mrow * APITCH + kc0 + 16]);
            v16h a2  = SHUF16(a2lo, a2hi);
            #pragma unroll
            for (int tj = 0; tj < 2; ++tj) {
                const int nrow = wn * 32 + tj * 16 + laneM;
                const int kb   = laneHi * 8;
                v8h b0 = *reinterpret_cast<const v8h*>(&Bs[cur][0][nrow * APITCH + kb]);
                v8h b1 = *reinterpret_cast<const v8h*>(&Bs[cur][0][nrow * APITCH + kb + 16]);
                v16h bY = SHUF16(b0, b1);
                acc[0][ti][tj] = __builtin_amdgcn_wmma_f32_16x16x32_f16(
                    false, a1, false, bY, (short)0, acc[0][ti][tj], false, false);
                v8h c0 = *reinterpret_cast<const v8h*>(&Bs[cur][1][nrow * APITCH + kb]);
                v8h c1 = *reinterpret_cast<const v8h*>(&Bs[cur][1][nrow * APITCH + kb + 16]);
                v16h bA = SHUF16(c0, c1);
                acc[1][ti][tj] = __builtin_amdgcn_wmma_f32_16x16x32_f16(
                    false, a2, false, bA, (short)0, acc[1][ti][tj], false, false);
            }
        }
        __syncthreads();
    }

    #pragma unroll
    for (int ti = 0; ti < 2; ++ti) {
        #pragma unroll
        for (int tj = 0; tj < 2; ++tj) {
            const int nn  = n0 + wn * 32 + tj * 16 + laneM;
            const float b0 = by[nn];
            const float b1 = bya[nn];
            #pragma unroll
            for (int r = 0; r < 8; ++r) {
                const int mm = m0 + wm * 32 + ti * 16 + laneHi * 8 + r;
                const float vy = acc[0][ti][tj][r] + b0;
                const float vg = acc[1][ti][tj][r] + b1;
                y[(size_t)mm * DD + nn] = vy * silu_f(vg);
            }
        }
    }
}

// ---------------------------------------------------------------------------
extern "C" void kernel_launch(void* const* d_in, const int* in_sizes, int n_in,
                              void* d_out, int out_size, void* d_ws, size_t ws_size,
                              hipStream_t stream) {
    const float* x   = (const float*)d_in[0];
    const float* Wz  = (const float*)d_in[1];
    const float* bz  = (const float*)d_in[2];
    const float* Wza = (const float*)d_in[3];
    const float* bza = (const float*)d_in[4];
    const float* Wy  = (const float*)d_in[5];
    const float* by  = (const float*)d_in[6];
    const float* Wya = (const float*)d_in[7];
    const float* bya = (const float*)d_in[8];
    const float* nw  = (const float*)d_in[9];
    const float* lcr = (const float*)d_in[10];
    const float* lci = (const float*)d_in[11];
    float* out = (float*)d_out;

    // workspace layout
    float* zh  = (float*)d_ws;                         // B*L*D f32 (z -> h in place)
    float* ir  = zh + (size_t)MT * DD;                 // B*L
    float* caR = ir + MT;                              // B*NCH*D
    float* caI = caR + (size_t)BB * NCH * DD;          // B*NCH*D
    _Float16* xh   = (_Float16*)(caI + (size_t)BB * NCH * DD);  // B*L*D f16
    _Float16* wzh  = xh + (size_t)MT * DD;             // D*D f16 each
    _Float16* wzah = wzh + (size_t)DD * DD;
    _Float16* wyh  = wzah + (size_t)DD * DD;
    _Float16* wyah = wyh + (size_t)DD * DD;

    const dim3 blk(256);

    // f32 -> f16 packing
    cvt_kernel<<<(MT * DD / 8 + 255) / 256, blk, 0, stream>>>(x, xh, MT * DD);
    cvt_kernel<<<(DD * DD / 8 + 255) / 256, blk, 0, stream>>>(Wz,  wzh,  DD * DD);
    cvt_kernel<<<(DD * DD / 8 + 255) / 256, blk, 0, stream>>>(Wza, wzah, DD * DD);
    cvt_kernel<<<(DD * DD / 8 + 255) / 256, blk, 0, stream>>>(Wy,  wyh,  DD * DD);
    cvt_kernel<<<(DD * DD / 8 + 255) / 256, blk, 0, stream>>>(Wya, wyah, DD * DD);

    const dim3 g1(DD / BN, MT / BM);   // 16 x 128
    proj_z_kernel<<<g1, blk, 0, stream>>>(xh, wzh, bz, wzah, bza, zh);

    scan_local_kernel<<<(BB * NCH * DD) / 256, blk, 0, stream>>>(zh, caR, caI);
    scan_carry_kernel<<<(BB * DD) / 256, blk, 0, stream>>>(caR, caI, lcr, lci);
    scan_apply_kernel<<<(BB * NCH * DD) / 256, blk, 0, stream>>>(zh, caR, caI);

    rms_kernel<<<MT / 8, blk, 0, stream>>>(zh, ir);

    out_gemm_kernel<<<g1, blk, 0, stream>>>(zh, xh, wyh, by, wyah, bya, nw, ir, out);
}
